// LesionMatchingModel_18915035972033
// MI455X (gfx1250) — matrix operand order repaired and checked by reading, not compiled, not run on hardware
//
#include <hip/hip_runtime.h>
#include <hip/hip_bf16.h>
#include <math.h>

typedef __attribute__((ext_vector_type(16))) _Float16 v16h;
typedef __attribute__((ext_vector_type(8)))  float    v8f;

#define DVOL   262144        // 64*64*64
#define NWIN   4096          // 16*16*16 windows
#define KPTS   512
#define NCH    64

// ---- output offsets (floats) ----
#define OUT_G1  0            // (2,1,1,512,3) = 3072
#define OUT_G2  3072
#define OUT_L1  6144         // (2,512) = 1024
#define OUT_L2  7168
#define OUT_SC  8192         // (2*512*512, 2) = 1048576
#define OUT_DN  1056768      // (2,512,512)   = 524288

// ---------------------------------------------------------------------------
// K1: per-4^3-window max of raw kpt values (sigmoid is monotone -> skip it).
// One thread per window; 4 (vol,b) volumes * 4096 windows = 16384 threads.
// ---------------------------------------------------------------------------
__global__ void winmax_kernel(const float* __restrict__ kpt1,
                              const float* __restrict__ kpt2,
                              float* __restrict__ cand_val,
                              int*   __restrict__ cand_idx) {
    int gid = blockIdx.x * blockDim.x + threadIdx.x;     // 0..16383
    if (gid >= 4 * NWIN) return;
    int vol = gid >> 13;
    int b   = (gid >> 12) & 1;
    int w   = gid & (NWIN - 1);
    const float* kp = (vol ? kpt2 : kpt1) + (size_t)b * DVOL;
    int wz = w >> 8, wy = (w >> 4) & 15, wx = w & 15;
    float best = -INFINITY;
    int   besti = 0;
    for (int dz = 0; dz < 4; ++dz) {
        int z = wz * 4 + dz;
        for (int dy = 0; dy < 4; ++dy) {
            int y = wy * 4 + dy;
            int base = (z * 64 + y) * 64 + wx * 4;
            #pragma unroll
            for (int dx = 0; dx < 4; ++dx) {
                float v = kp[base + dx];
                if (v > best) { best = v; besti = base + dx; } // first-occurrence tiebreak
            }
        }
    }
    cand_val[gid] = best;
    cand_idx[gid] = besti;
}

// ---------------------------------------------------------------------------
// K2: one block per (vol,b): bitonic sort of 4096 candidates in LDS,
// descending by value (ascending index on ties, matching jax top_k),
// emit top-512 grid coords + selected voxel indices.
// ---------------------------------------------------------------------------
__global__ void topk_kernel(const float* __restrict__ cand_val,
                            const int*   __restrict__ cand_idx,
                            int*   __restrict__ sel_idx,
                            float* __restrict__ d_out) {
    __shared__ float sv[NWIN];
    __shared__ int   si[NWIN];
    int vb = blockIdx.x;                  // vol*2 + b
    int t  = threadIdx.x;                 // 0..511
    for (int i = t; i < NWIN; i += 512) {
        sv[i] = cand_val[vb * NWIN + i];
        si[i] = cand_idx[vb * NWIN + i];
    }
    __syncthreads();
    for (int kk = 2; kk <= NWIN; kk <<= 1) {
        for (int jj = kk >> 1; jj > 0; jj >>= 1) {
            for (int i = t; i < NWIN; i += 512) {
                int p = i ^ jj;
                if (p > i) {
                    float vi = sv[i], vp = sv[p];
                    int   ii = si[i], ip = si[p];
                    // "before" in desired order: larger value first, then smaller index
                    bool before_p = (vp > vi) || (vp == vi && ip < ii);
                    bool sw = ((i & kk) == 0) ? before_p : !before_p;
                    if (sw) { sv[i] = vp; sv[p] = vi; si[i] = ip; si[p] = ii; }
                }
            }
            __syncthreads();
        }
    }
    // first 512 entries are the keypoints, in top_k order
    int vol = vb >> 1, b = vb & 1;
    int flat = si[t];
    sel_idx[vb * KPTS + t] = flat;
    int x = flat & 63, y = (flat >> 6) & 63, z = flat >> 12;
    float gx = (float)x * (2.0f / 63.0f) - 1.0f;
    float gy = (float)y * (2.0f / 63.0f) - 1.0f;
    float gz = (float)z * (2.0f / 63.0f) - 1.0f;
    float* g = d_out + (vol ? OUT_G2 : OUT_G1) + (size_t)(b * KPTS + t) * 3;
    g[0] = gx; g[1] = gy; g[2] = gz;
}

// ---------------------------------------------------------------------------
// K3: trilinear sample. One wave32 per keypoint: lane c and c+32 handle two
// channels of the feature volume; the kpt-map sample (logit) is uniform.
// Also produces per-point descriptor L2 norms via wave shuffle reduction.
// ---------------------------------------------------------------------------
__global__ void sample_kernel(const float* __restrict__ kpt1,
                              const float* __restrict__ kpt2,
                              const float* __restrict__ feat1,
                              const float* __restrict__ feat2,
                              const int*   __restrict__ sel_idx,
                              float* __restrict__ desc,
                              float* __restrict__ norms,
                              float* __restrict__ d_out) {
    int lane = threadIdx.x & 31;
    int wid  = threadIdx.x >> 5;
    int p    = blockIdx.x * 8 + wid;       // 0..2047  == (vol*2+b)*512 + i
    int vol  = p >> 10;
    int b    = (p >> 9) & 1;
    int flat = sel_idx[p];
    int x = flat & 63, y = (flat >> 6) & 63, z = flat >> 12;
    // replicate reference coord math exactly
    float gx = (float)x * (2.0f / 63.0f) - 1.0f;
    float gy = (float)y * (2.0f / 63.0f) - 1.0f;
    float gz = (float)z * (2.0f / 63.0f) - 1.0f;
    float ix = ((gx + 1.0f) * 64.0f - 1.0f) * 0.5f;
    float iy = ((gy + 1.0f) * 64.0f - 1.0f) * 0.5f;
    float iz = ((gz + 1.0f) * 64.0f - 1.0f) * 0.5f;
    float x0f = floorf(ix), y0f = floorf(iy), z0f = floorf(iz);
    int x0 = (int)x0f, y0 = (int)y0f, z0 = (int)z0f;
    float fx = ix - x0f, fy = iy - y0f, fz = iz - z0f;

    const float* kp = (vol ? kpt2 : kpt1) + (size_t)b * DVOL;
    const float* ft = (vol ? feat2 : feat1) + (size_t)b * NCH * DVOL;

    float acc0 = 0.0f, acc1 = 0.0f, accl = 0.0f;
    #pragma unroll
    for (int c8 = 0; c8 < 8; ++c8) {
        int dz = c8 >> 2, dy = (c8 >> 1) & 1, dx = c8 & 1;
        int xi = x0 + dx, yi = y0 + dy, zi = z0 + dz;
        if (xi < 0 || xi > 63 || yi < 0 || yi > 63 || zi < 0 || zi > 63) continue;
        float wgt = (dx ? fx : 1.0f - fx) * (dy ? fy : 1.0f - fy) * (dz ? fz : 1.0f - fz);
        int vox = (zi * 64 + yi) * 64 + xi;
        accl += wgt * kp[vox];
        acc0 += wgt * ft[(size_t)lane * DVOL + vox];
        acc1 += wgt * ft[(size_t)(lane + 32) * DVOL + vox];
    }
    desc[(size_t)p * NCH + lane]      = acc0;
    desc[(size_t)p * NCH + lane + 32] = acc1;
    float ss = acc0 * acc0 + acc1 * acc1;
    #pragma unroll
    for (int m = 16; m > 0; m >>= 1) ss += __shfl_xor(ss, m, 32);
    if (lane == 0) {
        norms[p] = sqrtf(ss);
        d_out[OUT_L1 + p] = accl;      // OUT_L1 + vol*1024 + b*512 + i == OUT_L1 + p
    }
}

// ---------------------------------------------------------------------------
// K4: build f16 GEMM operands. A_k = d1 * fc_w[k] (pre-scaled), B = d2.
// ---------------------------------------------------------------------------
__global__ void prep16_kernel(const float* __restrict__ desc,
                              const float* __restrict__ fc_w,
                              _Float16* __restrict__ A16,
                              _Float16* __restrict__ B16) {
    int gid = blockIdx.x * blockDim.x + threadIdx.x;   // 0..65535 = (b,i,c)
    int b = gid >> 15;
    int rem = gid & 32767;
    int i = rem >> 6, c = rem & 63;
    float d1v = desc[(size_t)((b      * KPTS + i)) * NCH + c];       // vol 0
    float d2v = desc[(size_t)(((2 + b) * KPTS + i)) * NCH + c];      // vol 1
    B16[(size_t)(b * KPTS + i) * NCH + c] = (_Float16)d2v;
    A16[(size_t)((b * 2 + 0) * KPTS + i) * NCH + c] = (_Float16)(d1v * fc_w[c]);
    A16[(size_t)((b * 2 + 1) * KPTS + i) * NCH + c] = (_Float16)(d1v * fc_w[64 + c]);
}

// ---------------------------------------------------------------------------
// K5: scores GEMM on v_wmma_f32_16x16x32_f16. One wave per 16x16 output tile,
// K=64 in two WMMA issues, f32 accumulate, bias added at store.
// A/B use contiguous k order (k = 16*(lane/16) + e, same in both); C/D per CDNA5 ISA (wave32).
// ---------------------------------------------------------------------------
__global__ void wmma_scores_kernel(const _Float16* __restrict__ A16,
                                   const _Float16* __restrict__ B16,
                                   const float* __restrict__ fcb,
                                   float* __restrict__ outsc) {
    int lane = threadIdx.x & 31;
    int wid  = threadIdx.x >> 5;
    int T    = blockIdx.x * 8 + wid;          // 0..4095 tiles
    int b    = T >> 11;
    int k    = (T >> 10) & 1;
    int tile = T & 1023;
    int ti   = tile >> 5, tj = tile & 31;

    const _Float16* Abase = A16 + (size_t)((b * 2 + k) * KPTS + ti * 16 + (lane & 15)) * NCH;
    const _Float16* Bbase = B16 + (size_t)(b * KPTS + tj * 16 + (lane & 15)) * NCH;
    int klo_a = (lane & 16) ? 16 : 0;   // A: lanes 16-31 hold K 16..31 of the step
    int klo_b = (lane & 16) ? 16 : 0;   // B: lanes 16-31 hold K 16..31 of the step

    v8f acc = {0.f, 0.f, 0.f, 0.f, 0.f, 0.f, 0.f, 0.f};
    #pragma unroll
    for (int k0 = 0; k0 < NCH; k0 += 32) {
        v16h af, bf;
        #pragma unroll
        for (int e = 0; e < 16; ++e) {
            af[e] = Abase[k0 + klo_a + e];
            bf[e] = Bbase[k0 + klo_b + e];
        }
        acc = __builtin_amdgcn_wmma_f32_16x16x32_f16(
            /*neg_a=*/false, af, /*neg_b=*/false, bf,
            /*c_mod=*/(short)0, acc, /*reuse_a=*/false, /*reuse_b=*/false);
    }
    float bias = fcb[k];
    int j = tj * 16 + (lane & 15);
    int ibase = ti * 16 + ((lane & 16) ? 8 : 0);
    #pragma unroll
    for (int r = 0; r < 8; ++r) {
        int i = ibase + r;
        outsc[(((size_t)(b * KPTS + i) * KPTS + j) << 1) + k] = acc[r] + bias;
    }
}

// ---------------------------------------------------------------------------
// K6: desc_norm closed form: since the reference uses `a` as numerator for
// both normalized tensors, diff_c = a_c*(inv1_i - inv2_j), so
// ||diff|| = n1_i * |inv1_i - inv2_j|.
// ---------------------------------------------------------------------------
__global__ void descnorm_kernel(const float* __restrict__ norms,
                                float* __restrict__ outdn) {
    int gid = blockIdx.x * blockDim.x + threadIdx.x;   // 0..524287
    int b  = gid >> 18;
    int ij = gid & 262143;
    int i  = ij >> 9, j = ij & 511;
    float n1 = norms[b * KPTS + i];          // vol 0
    float n2 = norms[(2 + b) * KPTS + j];    // vol 1
    float inv1 = 1.0f / (1e-6f + n1);
    float inv2 = 1.0f / (1e-6f + n2);
    outdn[gid] = n1 * fabsf(inv1 - inv2);
}

// ---------------------------------------------------------------------------
extern "C" void kernel_launch(void* const* d_in, const int* in_sizes, int n_in,
                              void* d_out, int out_size, void* d_ws, size_t ws_size,
                              hipStream_t stream) {
    const float* kpt1  = (const float*)d_in[0];
    const float* kpt2  = (const float*)d_in[1];
    const float* feat1 = (const float*)d_in[2];
    const float* feat2 = (const float*)d_in[3];
    const float* fc_w  = (const float*)d_in[4];
    const float* fc_b  = (const float*)d_in[5];
    float* out = (float*)d_out;

    // workspace layout (bytes)
    char* ws = (char*)d_ws;
    float*    cand_val = (float*)(ws + 0);         //  4*4096*4 = 64KB
    int*      cand_idx = (int*)  (ws + 65536);     //  64KB
    int*      sel_idx  = (int*)  (ws + 131072);    //  4*512*4  = 8KB
    float*    norms    = (float*)(ws + 139264);    //  8KB
    float*    desc     = (float*)(ws + 147456);    //  4*512*64*4 = 512KB
    _Float16* A16      = (_Float16*)(ws + 671744); //  2*2*512*64*2 = 256KB
    _Float16* B16      = (_Float16*)(ws + 933888); //  2*512*64*2 = 128KB
    // total: 1,064,960 bytes

    winmax_kernel<<<64, 256, 0, stream>>>(kpt1, kpt2, cand_val, cand_idx);
    topk_kernel<<<4, 512, 0, stream>>>(cand_val, cand_idx, sel_idx, out);
    sample_kernel<<<256, 256, 0, stream>>>(kpt1, kpt2, feat1, feat2,
                                           sel_idx, desc, norms, out);
    prep16_kernel<<<256, 256, 0, stream>>>(desc, fc_w, A16, B16);
    wmma_scores_kernel<<<512, 256, 0, stream>>>(A16, B16, fc_b, out + OUT_SC);
    descnorm_kernel<<<2048, 256, 0, stream>>>(norms, out + OUT_DN);
}
